// GINEdgeLayer_48129403519264
// MI455X (gfx1250) — matrix operand order repaired and checked by reading, not compiled, not run on hardware
//
#include <hip/hip_runtime.h>

typedef _Float16 h16;
typedef __attribute__((ext_vector_type(16))) _Float16 v16h;
typedef __attribute__((ext_vector_type(8)))  float    v8f;

union FragU { v16h h; unsigned u[8]; };

__device__ __forceinline__ unsigned lds_u32(const h16* p) {
    return *(const unsigned*)p;
}

// A-matrix (16-bit, 16x32) K offset for reg v, lane-group g (ISA 7.12.2 table)
__device__ __forceinline__ constexpr int kmapA(int v, int g) {
    return ((v & 4) << 2) + (g << 3) + ((v & 3) << 1);
}
// B-matrix (16-bit, 32x16) K offset: same K order as A (ISA 7.12.2 table)
__device__ __forceinline__ constexpr int kmapB(int v, int g) {
    return ((v & 4) << 2) + (g << 3) + ((v & 3) << 1);
}

__device__ __forceinline__ v8f wmma16(v16h a, v16h b, v8f c) {
    return __builtin_amdgcn_wmma_f32_16x16x32_f16(false, a, false, b, (short)0, c,
                                                  false, false);
}

// ---------------------------------------------------------------------------
// Kernel 0: agg = (1 + eps) * H
// ---------------------------------------------------------------------------
__global__ void init_agg_kernel(const float* __restrict__ H,
                                const float* __restrict__ eps,
                                float* __restrict__ agg, int n) {
    int i = blockIdx.x * blockDim.x + threadIdx.x;
    if (i < n) agg[i] = (1.0f + eps[0]) * H[i];
}

// ---------------------------------------------------------------------------
// Kernel 1: per-edge message MLP (192->256 relu ->128) + atomic scatter.
// 8 waves/block; each wave owns a 32-edge tile (two 16-row WMMA tiles) per
// iteration, so every B fragment read from LDS feeds TWO v_wmma ops.
// Input staging and hidden staging alias one per-wave LDS buffer (the whole
// A tile is register-resident in fragments before any hidden write).
// ---------------------------------------------------------------------------
__global__ __launch_bounds__(256, 1)
void edge_msg_kernel(const float* __restrict__ H,
                     const int*   __restrict__ src,
                     const int*   __restrict__ dst,
                     const float* __restrict__ EA,
                     const float* __restrict__ W1,   // [192][256]
                     const float* __restrict__ b1,   // [256]
                     const float* __restrict__ W2,   // [256][128]
                     const float* __restrict__ b2,   // [128]
                     float* __restrict__ agg,
                     int n_tiles) {                  // tiles of 32 edges
    constexpr int W1S = 196;  // W1^T row stride (K=192 padded, conflict-free)
    constexpr int W2S = 260;  // W2^T row stride (K=256 padded)
    constexpr int AS  = 196;  // input staging stride (192 padded)
    constexpr int HS  = 260;  // hidden staging stride (256 padded)

    __shared__ h16   sW1t[256 * W1S];     // 100,352 B
    __shared__ h16   sW2t[128 * W2S];     //  66,560 B
    __shared__ h16   sBuf[8][32 * HS];    // 133,120 B (input & hidden, aliased)
    __shared__ float sB1[256];
    __shared__ float sB2[128];
    __shared__ int   sDst[8][32];

    const int tid = threadIdx.x;
    for (int i = tid; i < 192 * 256; i += 256) {
        int k = i >> 8, n = i & 255;
        sW1t[n * W1S + k] = (h16)W1[i];
    }
    for (int i = tid; i < 256 * 128; i += 256) {
        int k = i >> 7, n = i & 127;
        sW2t[n * W2S + k] = (h16)W2[i];
    }
    sB1[tid] = b1[tid];
    if (tid < 128) sB2[tid] = b2[tid];
    __syncthreads();

    const int w = tid >> 5, lane = tid & 31;
    const int n16 = lane & 15, g = lane >> 4;
    h16* Buf  = sBuf[w];
    int* dstw = sDst[w];

    for (int tile = blockIdx.x * 8 + w; tile < n_tiles; tile += gridDim.x * 8) {
        const int e0 = tile << 5;

        // ---- stage A = [H[src[e]] | edge_attr[e]] as f16, 32 x 192 --------
        #pragma unroll 4
        for (int m = 0; m < 32; ++m) {
            const int s = src[e0 + m];                  // wave-uniform -> s_load
            const float* hr = H  + (size_t)s * 128;
            const float* er = EA + (size_t)(e0 + m) * 64;
            #pragma unroll
            for (int c = 0; c < 4; ++c)
                Buf[m * AS + c * 32 + lane] = (h16)hr[c * 32 + lane];
            #pragma unroll
            for (int c = 0; c < 2; ++c)
                Buf[m * AS + 128 + c * 32 + lane] = (h16)er[c * 32 + lane];
        }
        dstw[lane] = dst[e0 + lane];
        asm volatile("s_wait_dscnt 0" ::: "memory");

        // ---- layer 1: [32x192] @ [192x256] + b1, relu ---------------------
        FragU a1[2][6];
        #pragma unroll
        for (int t = 0; t < 2; ++t)
            #pragma unroll
            for (int kt = 0; kt < 6; ++kt)
                #pragma unroll
                for (int v = 0; v < 8; ++v)
                    a1[t][kt].u[v] =
                        lds_u32(Buf + (t * 16 + n16) * AS + kt * 32 + kmapA(v, g));

        #pragma unroll 1
        for (int nt = 0; nt < 16; ++nt) {
            const float bv = sB1[nt * 16 + n16];
            v8f acc[2];
            #pragma unroll
            for (int t = 0; t < 2; ++t)
                #pragma unroll
                for (int r = 0; r < 8; ++r) acc[t][r] = bv;
            const h16* wrow = sW1t + (nt * 16 + n16) * W1S;
            #pragma unroll
            for (int kt = 0; kt < 6; ++kt) {
                FragU bf;
                #pragma unroll
                for (int v = 0; v < 8; ++v)
                    bf.u[v] = lds_u32(wrow + kt * 32 + kmapB(v, g));
                acc[0] = wmma16(a1[0][kt].h, bf.h, acc[0]);
                acc[1] = wmma16(a1[1][kt].h, bf.h, acc[1]);
            }
            #pragma unroll
            for (int t = 0; t < 2; ++t)
                #pragma unroll
                for (int r = 0; r < 8; ++r)
                    Buf[(t * 16 + g * 8 + r) * HS + nt * 16 + n16] =
                        (h16)__builtin_fmaxf(acc[t][r], 0.0f);
        }
        asm volatile("s_wait_dscnt 0" ::: "memory");

        // ---- layer 2: [32x256] @ [256x128] + b2, atomic scatter -----------
        FragU a2[2][8];
        #pragma unroll
        for (int t = 0; t < 2; ++t)
            #pragma unroll
            for (int kt = 0; kt < 8; ++kt)
                #pragma unroll
                for (int v = 0; v < 8; ++v)
                    a2[t][kt].u[v] =
                        lds_u32(Buf + (t * 16 + n16) * HS + kt * 32 + kmapA(v, g));

        #pragma unroll 1
        for (int nt = 0; nt < 8; ++nt) {
            const float bv = sB2[nt * 16 + n16];
            v8f acc[2];
            #pragma unroll
            for (int t = 0; t < 2; ++t)
                #pragma unroll
                for (int r = 0; r < 8; ++r) acc[t][r] = bv;
            const h16* wrow = sW2t + (nt * 16 + n16) * W2S;
            #pragma unroll
            for (int kt = 0; kt < 8; ++kt) {
                FragU bf;
                #pragma unroll
                for (int v = 0; v < 8; ++v)
                    bf.u[v] = lds_u32(wrow + kt * 32 + kmapB(v, g));
                acc[0] = wmma16(a2[0][kt].h, bf.h, acc[0]);
                acc[1] = wmma16(a2[1][kt].h, bf.h, acc[1]);
            }
            #pragma unroll
            for (int t = 0; t < 2; ++t)
                #pragma unroll
                for (int r = 0; r < 8; ++r) {
                    const int m = t * 16 + g * 8 + r;
                    unsafeAtomicAdd(agg + (size_t)dstw[m] * 128 + nt * 16 + n16,
                                    acc[t][r]);
                }
        }
    }
}

// ---------------------------------------------------------------------------
// Kernel 2: per-node self MLP (128->256 relu ->128) on agg, write d_out.
// Same M=32-per-wave structure.
// ---------------------------------------------------------------------------
__global__ __launch_bounds__(256, 1)
void node_mlp_kernel(const float* __restrict__ X,    // agg [N][128]
                     const float* __restrict__ W1,   // [128][256]
                     const float* __restrict__ b1,   // [256]
                     const float* __restrict__ W2,   // [256][128]
                     const float* __restrict__ b2,   // [128]
                     float* __restrict__ out,
                     int n_tiles) {                  // tiles of 32 nodes
    constexpr int W1S = 132;  // K=128 padded
    constexpr int W2S = 260;  // K=256 padded
    constexpr int AS  = 132;
    constexpr int HS  = 260;

    __shared__ h16   sW1t[256 * W1S];     // 67,584 B
    __shared__ h16   sW2t[128 * W2S];     // 66,560 B
    __shared__ h16   sBuf[8][32 * HS];    // 133,120 B (input & hidden, aliased)
    __shared__ float sB1[256];
    __shared__ float sB2[128];

    const int tid = threadIdx.x;
    for (int i = tid; i < 128 * 256; i += 256) {
        int k = i >> 8, n = i & 255;
        sW1t[n * W1S + k] = (h16)W1[i];
    }
    for (int i = tid; i < 256 * 128; i += 256) {
        int k = i >> 7, n = i & 127;
        sW2t[n * W2S + k] = (h16)W2[i];
    }
    sB1[tid] = b1[tid];
    if (tid < 128) sB2[tid] = b2[tid];
    __syncthreads();

    const int w = tid >> 5, lane = tid & 31;
    const int n16 = lane & 15, g = lane >> 4;
    h16* Buf = sBuf[w];

    for (int tile = blockIdx.x * 8 + w; tile < n_tiles; tile += gridDim.x * 8) {
        const int r0 = tile << 5;

        #pragma unroll 4
        for (int m = 0; m < 32; ++m) {
            const float* xr = X + (size_t)(r0 + m) * 128;
            #pragma unroll
            for (int c = 0; c < 4; ++c)
                Buf[m * AS + c * 32 + lane] = (h16)xr[c * 32 + lane];
        }
        asm volatile("s_wait_dscnt 0" ::: "memory");

        // layer 1: K = 128 -> 4 k-tiles
        FragU a1[2][4];
        #pragma unroll
        for (int t = 0; t < 2; ++t)
            #pragma unroll
            for (int kt = 0; kt < 4; ++kt)
                #pragma unroll
                for (int v = 0; v < 8; ++v)
                    a1[t][kt].u[v] =
                        lds_u32(Buf + (t * 16 + n16) * AS + kt * 32 + kmapA(v, g));

        #pragma unroll 1
        for (int nt = 0; nt < 16; ++nt) {
            const float bv = sB1[nt * 16 + n16];
            v8f acc[2];
            #pragma unroll
            for (int t = 0; t < 2; ++t)
                #pragma unroll
                for (int r = 0; r < 8; ++r) acc[t][r] = bv;
            const h16* wrow = sW1t + (nt * 16 + n16) * W1S;
            #pragma unroll
            for (int kt = 0; kt < 4; ++kt) {
                FragU bf;
                #pragma unroll
                for (int v = 0; v < 8; ++v)
                    bf.u[v] = lds_u32(wrow + kt * 32 + kmapB(v, g));
                acc[0] = wmma16(a1[0][kt].h, bf.h, acc[0]);
                acc[1] = wmma16(a1[1][kt].h, bf.h, acc[1]);
            }
            #pragma unroll
            for (int t = 0; t < 2; ++t)
                #pragma unroll
                for (int r = 0; r < 8; ++r)
                    Buf[(t * 16 + g * 8 + r) * HS + nt * 16 + n16] =
                        (h16)__builtin_fmaxf(acc[t][r], 0.0f);
        }
        asm volatile("s_wait_dscnt 0" ::: "memory");

        // layer 2: K = 256 -> 8 k-tiles, write output directly
        FragU a2[2][8];
        #pragma unroll
        for (int t = 0; t < 2; ++t)
            #pragma unroll
            for (int kt = 0; kt < 8; ++kt)
                #pragma unroll
                for (int v = 0; v < 8; ++v)
                    a2[t][kt].u[v] =
                        lds_u32(Buf + (t * 16 + n16) * HS + kt * 32 + kmapA(v, g));

        #pragma unroll 1
        for (int nt = 0; nt < 8; ++nt) {
            const float bv = sB2[nt * 16 + n16];
            v8f acc[2];
            #pragma unroll
            for (int t = 0; t < 2; ++t)
                #pragma unroll
                for (int r = 0; r < 8; ++r) acc[t][r] = bv;
            const h16* wrow = sW2t + (nt * 16 + n16) * W2S;
            #pragma unroll
            for (int kt = 0; kt < 8; ++kt) {
                FragU bf;
                #pragma unroll
                for (int v = 0; v < 8; ++v)
                    bf.u[v] = lds_u32(wrow + kt * 32 + kmapB(v, g));
                acc[0] = wmma16(a2[0][kt].h, bf.h, acc[0]);
                acc[1] = wmma16(a2[1][kt].h, bf.h, acc[1]);
            }
            #pragma unroll
            for (int t = 0; t < 2; ++t)
                #pragma unroll
                for (int r = 0; r < 8; ++r)
                    out[(size_t)(r0 + t * 16 + g * 8 + r) * 128 + nt * 16 + n16] =
                        acc[t][r];
        }
    }
}

// ---------------------------------------------------------------------------
extern "C" void kernel_launch(void* const* d_in, const int* in_sizes, int n_in,
                              void* d_out, int out_size, void* d_ws, size_t ws_size,
                              hipStream_t stream) {
    const float* H   = (const float*)d_in[0];
    const int*   ei  = (const int*)d_in[1];
    const float* EA  = (const float*)d_in[2];
    const float* eps = (const float*)d_in[3];
    const float* mW1 = (const float*)d_in[4];
    const float* mb1 = (const float*)d_in[5];
    const float* mW2 = (const float*)d_in[6];
    const float* mb2 = (const float*)d_in[7];
    const float* sW1 = (const float*)d_in[8];
    const float* sb1 = (const float*)d_in[9];
    const float* sW2 = (const float*)d_in[10];
    const float* sb2 = (const float*)d_in[11];
    float* out = (float*)d_out;
    float* agg = (float*)d_ws;   // N*128 floats = 10.24 MB scratch

    const int N_ = in_sizes[0] / 128;     // 20000
    const int E_ = in_sizes[1] / 2;       // 200000 (edge_index is [2][E])
    const int* src = ei;
    const int* dst = ei + E_;

    const int ne = N_ * 128;
    init_agg_kernel<<<(ne + 255) / 256, 256, 0, stream>>>(H, eps, agg, ne);

    const int etiles = E_ / 32;           // 6250 (E divisible by 32)
    edge_msg_kernel<<<256, 256, 0, stream>>>(H, src, dst, EA,
                                             mW1, mb1, mW2, mb2, agg, etiles);

    const int ntiles = N_ / 32;           // 625
    node_mlp_kernel<<<80, 256, 0, stream>>>(agg, sW1, sb1, sW2, sb2, out, ntiles);
}
